// LearnedTokenPooler_30648886624911
// MI455X (gfx1250) — matrix operand order repaired and checked by reading, not compiled, not run on hardware
//
#include <hip/hip_runtime.h>

typedef __attribute__((ext_vector_type(16))) __bf16 v16bf;
typedef __attribute__((ext_vector_type(8)))  __bf16 v8bf;
typedef __attribute__((ext_vector_type(4)))  __bf16 v4bf;
typedef __attribute__((ext_vector_type(8)))  float  v8f;

namespace {

constexpr int kB = 8, kN = 32768, kC = 512, kS = 256;
constexpr int NCHUNK = 16;            // split-N factor (grid.x)
constexpr int SBLK   = 128;           // query rows per workgroup (x read only 2x total)
constexpr int WAVES  = 8;             // 16 query rows per wave
constexpr int KT     = 32;            // keys per tile (= one WMMA K-step for PV)
constexpr int SQP    = 528;           // sQ row stride (halves): 32B-mult, 264w -> 2-way banks
constexpr int SXP    = 520;           // sX row stride (halves): 16B-mult, 260w -> conflict-free
constexpr int KTP    = 40;            // sXT row stride (halves): 16B-mult, 20w -> conflict-free reads
constexpr int PTP    = 48;            // sP row stride (halves): 32B-mult, 24w -> 2-way
constexpr int OSTR   = 260;           // outbuf row stride (floats): 16B-mult, conflict-free
constexpr int NPW    = kN / NCHUNK;   // 2048 keys per workgroup
constexpr int TILES  = NPW / KT;      // 64
constexpr int NPART  = kB * (kS / SBLK) * NCHUNK;   // 256 partials (~67.4 MB workspace)

// LDS carve-up (all offsets 128B aligned); outbuf aliases sQ after the tile loop.
constexpr unsigned OFF_Q   = 0;                                  // 128*528*2 = 135168
constexpr unsigned OFF_X0  = OFF_Q   + SBLK * SQP * 2;           // 32*520*2  = 33280
constexpr unsigned OFF_X1  = OFF_X0  + KT * SXP * 2;
constexpr unsigned OFF_XT0 = OFF_X1  + KT * SXP * 2;             // 512*40*2  = 40960
constexpr unsigned OFF_XT1 = OFF_XT0 + kC * KTP * 2;
constexpr unsigned OFF_P   = OFF_XT1 + kC * KTP * 2;             // 8*16*48*2 = 12288
constexpr unsigned LDS_TOT = OFF_P + WAVES * 16 * PTP * 2;       // 295936 B <= 320KB
static_assert(LDS_TOT <= 320u * 1024u, "LDS budget");
static_assert(WAVES * 16 * OSTR * 4 <= SBLK * SQP * 2, "outbuf fits in sQ region");

__device__ __forceinline__ __bf16 bf16rne(float f) {
  unsigned u = __builtin_bit_cast(unsigned, f);
  u += 0x7FFFu + ((u >> 16) & 1u);
  return __builtin_bit_cast(__bf16, (unsigned short)(u >> 16));
}

// A-fragment (16xK, 16-bit): lane = row; halves 0..7 -> K=k0+8*hi+h, halves 8..15 -> K=k0+16+8*hi+(h-8)
__device__ __forceinline__ v16bf loadA(const __bf16* p0, int stride, int k0, int lo, int hi) {
  const __bf16* p = p0 + lo * stride + k0 + 8 * hi;
  v8bf a = *reinterpret_cast<const v8bf*>(p);
  v8bf b = *reinterpret_cast<const v8bf*>(p + 16);
  return __builtin_shufflevector(a, b, 0,1,2,3,4,5,6,7,8,9,10,11,12,13,14,15);
}
// B-fragment (Kx16, 16-bit): lane = column; halves 0..7 -> K=k0+8*hi+h, halves 8..15 -> K=k0+16+8*hi+(h-8)
__device__ __forceinline__ v16bf loadB(const __bf16* p0, int stride, int k0, int lo, int hi) {
  const __bf16* p = p0 + lo * stride + k0 + 8 * hi;
  v8bf a = *reinterpret_cast<const v8bf*>(p);
  v8bf b = *reinterpret_cast<const v8bf*>(p + 16);
  return __builtin_shufflevector(a, b, 0,1,2,3,4,5,6,7,8,9,10,11,12,13,14,15);
}

} // namespace

__global__ __launch_bounds__(WAVES * 32, 1)
void flash_partial(const float* __restrict__ x, const float* __restrict__ q,
                   float* __restrict__ wsAcc, float* __restrict__ wsM,
                   float* __restrict__ wsL)
{
  __shared__ __align__(128) unsigned char smem[LDS_TOT];
  __bf16* sQ   = reinterpret_cast<__bf16*>(smem + OFF_Q);
  __bf16* sX0  = reinterpret_cast<__bf16*>(smem + OFF_X0);
  __bf16* sX1  = reinterpret_cast<__bf16*>(smem + OFF_X1);
  __bf16* sXT0 = reinterpret_cast<__bf16*>(smem + OFF_XT0);
  __bf16* sXT1 = reinterpret_cast<__bf16*>(smem + OFF_XT1);
  __bf16* sP   = reinterpret_cast<__bf16*>(smem + OFF_P);

  const int tid  = threadIdx.x;
  const int lane = tid & 31;
  const int wave = tid >> 5;
  const int lo   = lane & 15;
  const int hi   = lane >> 4;
  const int chunk = blockIdx.x, sh = blockIdx.y, b = blockIdx.z;
  const int srow0 = sh * SBLK;
  const int n0 = chunk * NPW;
  const float QSCALE = 0.04419417382415922f * 1.4426950408889634f;  // C^-1/2 * log2(e)

  const float* xbase = x + ((size_t)b * kN + n0) * kC;

  // --- prefetch tile 0 into registers (16 float4 / thread) ---
  float4 pf[16];
  {
    const float4* g = reinterpret_cast<const float4*>(xbase);
    #pragma unroll
    for (int n = 0; n < 16; ++n) pf[n] = g[tid + n * 256];
  }
  // --- stage scaled Q block (bf16, padded stride) ---
  for (int i = tid; i < SBLK * kC / 4; i += WAVES * 32) {
    float4 v = reinterpret_cast<const float4*>(q + srow0 * kC)[i];
    int row = i >> 7, c4 = (i & 127) << 2;
    __bf16* d = sQ + row * SQP + c4;
    d[0] = bf16rne(v.x * QSCALE); d[1] = bf16rne(v.y * QSCALE);
    d[2] = bf16rne(v.z * QSCALE); d[3] = bf16rne(v.w * QSCALE);
  }

  const v8f vzero = {0.f,0.f,0.f,0.f,0.f,0.f,0.f,0.f};
  v8f acc[kC / 16];
  #pragma unroll
  for (int cs = 0; cs < kC / 16; ++cs) acc[cs] = vzero;
  float mrun = -1e30f, lrun = 0.f;          // per-lane: this lane's s = lo

  __bf16* pw = sP + wave * 16 * PTP;
  const __bf16* qb = sQ + wave * 16 * SQP;
  const int c4f = (tid & 127) << 2;         // convert-phase column, fixed per thread
  const int kbs = tid >> 7;                 // convert-phase key base (0/1)

  for (int t = 0; t < TILES; ++t) {
    __bf16* bx  = (t & 1) ? sX1  : sX0;
    __bf16* bxt = (t & 1) ? sXT1 : sXT0;

    // ---- convert prefetched tile into both LDS layouts ----
    #pragma unroll
    for (int n = 0; n < 16; ++n) {
      const int key = kbs + 2 * n;
      float4 v = pf[n];
      __bf16 h0 = bf16rne(v.x), h1 = bf16rne(v.y), h2 = bf16rne(v.z), h3 = bf16rne(v.w);
      v4bf t4; t4.x = h0; t4.y = h1; t4.z = h2; t4.w = h3;
      *reinterpret_cast<v4bf*>(bx + key * SXP + c4f) = t4;
      bxt[(c4f + 0) * KTP + key] = h0;
      bxt[(c4f + 1) * KTP + key] = h1;
      bxt[(c4f + 2) * KTP + key] = h2;
      bxt[(c4f + 3) * KTP + key] = h3;
    }
    // ---- issue global prefetch for t+1 (latency hides behind compute of t) ----
    if (t + 1 < TILES) {
      const float4* g = reinterpret_cast<const float4*>(xbase + (size_t)(t + 1) * KT * kC);
      #pragma unroll
      for (int n = 0; n < 16; ++n) pf[n] = g[tid + n * 256];
    }
    __syncthreads();  // single barrier per tile (double-buffered sX/sXT)

    // ---- logits^T: lt (key, s) = X(key,c) x Q^T(c,s), K = C in 16 steps ----
    v8f lt[2] = { vzero, vzero };
    #pragma unroll
    for (int k = 0; k < kC / 32; ++k) {
      v16bf bq = loadB(qb, SQP, k * 32, lo, hi);
      #pragma unroll
      for (int nt = 0; nt < 2; ++nt) {
        v16bf ax = loadA(bx + nt * 16 * SXP, SXP, k * 32, lo, hi);
        lt[nt] = __builtin_amdgcn_wmma_f32_16x16x32_bf16(false, ax, false, bq,
                                                         (short)0, lt[nt], false, false);
      }
    }

    // ---- online softmax over this tile's 32 keys (per lane: s = lo) ----
    float tmax = -1e30f;
    #pragma unroll
    for (int nt = 0; nt < 2; ++nt)
      #pragma unroll
      for (int r = 0; r < 8; ++r) tmax = fmaxf(tmax, lt[nt][r]);
    tmax = fmaxf(tmax, __shfl_xor(tmax, 16, 32));
    const float mnew = fmaxf(mrun, tmax);
    const float sc = __builtin_amdgcn_exp2f(mrun - mnew);
    float rs = 0.f;
    #pragma unroll
    for (int nt = 0; nt < 2; ++nt)
      #pragma unroll
      for (int r = 0; r < 8; ++r) {
        float p = __builtin_amdgcn_exp2f(lt[nt][r] - mnew);
        lt[nt][r] = p;
        rs += p;
      }
    rs += __shfl_xor(rs, 16, 32);
    lrun = lrun * sc + rs;
    if (mnew > mrun) {                 // rescale rarely fires once the max stabilizes
      #pragma unroll
      for (int cs = 0; cs < kC / 16; ++cs) acc[cs] *= sc;
    }
    mrun = mnew;

    // ---- P -> LDS as (s x key) ----
    #pragma unroll
    for (int nt = 0; nt < 2; ++nt)
      #pragma unroll
      for (int r = 0; r < 8; ++r)
        pw[lo * PTP + nt * 16 + r + 8 * hi] = bf16rne(lt[nt][r]);

    // ---- context^T: acc(c,s) += X^T(c x 32key) x P(32key x s), one K-step ----
    v16bf bp = loadB(pw, PTP, 0, lo, hi);
    #pragma unroll
    for (int cs = 0; cs < kC / 16; ++cs) {
      v16bf axt = loadA(bxt + cs * 16 * KTP, KTP, 0, lo, hi);
      acc[cs] = __builtin_amdgcn_wmma_f32_16x16x32_bf16(false, axt, false, bp,
                                                        (short)0, acc[cs], false, false);
    }
  }

  // ---- partial writeout via LDS staging + async store-from-LDS (sQ region is dead) ----
  __syncthreads();
  const int part = (b * (kS / SBLK) + sh) * NCHUNK + chunk;
  float* ow = reinterpret_cast<float*>(smem) + wave * 16 * OSTR;
  const unsigned ldsBase = (unsigned)(unsigned long long)(uintptr_t)ow;
  const unsigned long long wsBase = (unsigned long long)(uintptr_t)wsAcc;
  #pragma unroll
  for (int h = 0; h < 2; ++h) {
    #pragma unroll
    for (int cs2 = 0; cs2 < 16; ++cs2) {
      const int cs = h * 16 + cs2;
      #pragma unroll
      for (int r = 0; r < 8; ++r)
        ow[lo * OSTR + cs2 * 16 + r + 8 * hi] = acc[cs][r];   // (s,c) rows, conflict-free stride
    }
    asm volatile("s_wait_dscnt 0x0" ::: "memory");            // LDS visible to async engine
    const unsigned gbase = (unsigned)((((unsigned)part * SBLK + wave * 16) * kC + h * 256) * 4u);
    for (int it = 0; it < 32; ++it) {                          // 16 rows x 256 floats, b128 chunks
      unsigned idx = (unsigned)(it * 32 + lane);
      unsigned row = idx >> 6, q16 = idx & 63;
      unsigned voff  = gbase + (row * kC + q16 * 4) * 4u;
      unsigned laddr = ldsBase + (row * OSTR + q16 * 4) * 4u;
      asm volatile("global_store_async_from_lds_b128 %0, %1, %2"
                   :: "v"(voff), "v"(laddr), "s"(wsBase) : "memory");
    }
    asm volatile("s_wait_asynccnt 0x0" ::: "memory");         // buffer reusable / done
  }
  if (hi == 0) {
    wsM[part * SBLK + wave * 16 + lo] = mrun;
    wsL[part * SBLK + wave * 16 + lo] = lrun;
  }
}

__global__ __launch_bounds__(128)
void combine_partials(const float* __restrict__ wsAcc, const float* __restrict__ wsM,
                      const float* __restrict__ wsL, float* __restrict__ out)
{
  const int bs = blockIdx.x;
  const int b = bs >> 8;
  const int s = bs & (kS - 1);
  const int shn = s / SBLK;
  const int sl = s - shn * SBLK;
  const int base = (b * (kS / SBLK) + shn) * NCHUNK;

  float m[NCHUNK], w[NCHUNK];
  float M = -1e30f;
  #pragma unroll
  for (int i = 0; i < NCHUNK; ++i) {
    m[i] = wsM[(base + i) * SBLK + sl];
    M = fmaxf(M, m[i]);
  }
  float denom = 0.f;
  #pragma unroll
  for (int i = 0; i < NCHUNK; ++i) {
    w[i] = __builtin_amdgcn_exp2f(m[i] - M);
    denom += w[i] * wsL[(base + i) * SBLK + sl];
  }
  const float inv = 1.0f / denom;
  for (int c = threadIdx.x; c < kC; c += 128) {
    float a = 0.f;
    #pragma unroll
    for (int i = 0; i < NCHUNK; ++i)
      a += w[i] * wsAcc[((size_t)(base + i) * SBLK + sl) * kC + c];
    out[((size_t)b * kS + s) * kC + c] = a * inv;
  }
}

extern "C" void kernel_launch(void* const* d_in, const int* /*in_sizes*/, int /*n_in*/,
                              void* d_out, int /*out_size*/, void* d_ws, size_t /*ws_size*/,
                              hipStream_t stream) {
  const float* x = (const float*)d_in[0];        // (B, N, C) fp32
  const float* q = (const float*)d_in[1];        // (S, C)    fp32
  float* out = (float*)d_out;                    // (B, S, C) fp32
  float* wsAcc = (float*)d_ws;
  float* wsM = wsAcc + (size_t)NPART * SBLK * kC;
  float* wsL = wsM + (size_t)NPART * SBLK;

  dim3 grid(NCHUNK, kS / SBLK, kB);              // 256 workgroups
  flash_partial<<<grid, WAVES * 32, 0, stream>>>(x, q, wsAcc, wsM, wsL);
  combine_partials<<<kB * kS, 128, 0, stream>>>(wsAcc, wsM, wsL, out);
}
